// SimplerXEyTransformerLayer_78434692759838
// MI455X (gfx1250) — matrix operand order repaired and checked by reading, not compiled, not run on hardware
//
#include <hip/hip_runtime.h>
#include <math.h>

// ---------------- problem constants ----------------
#define NB_   512   // sequence length n
#define DX_   128
#define DE_   64
#define DY_   64
#define H_    8
#define DF_   16

typedef __attribute__((ext_vector_type(16))) __bf16 v16bf;
typedef __attribute__((ext_vector_type(8)))  float  v8f;
typedef __attribute__((ext_vector_type(4)))  unsigned int u32x4;
typedef __attribute__((ext_vector_type(8)))  unsigned int u32x8;

// ---------------- workspace layout (float units) ----------------
static constexpr size_t OFF_Q   = 0;
static constexpr size_t OFF_K   = 131072;
static constexpr size_t OFF_V   = 262144;
static constexpr size_t OFF_W   = 393216;            // 2,097,152 floats
static constexpr size_t OFF_YE1 = 2490368;
static constexpr size_t OFF_YE2 = 2490624;
static constexpr size_t OFF_YX1 = 2490880;
static constexpr size_t OFF_YX2 = 2491136;
static constexpr size_t OFF_SXQ = 2491392;           // 2*512
static constexpr size_t OFF_SXK = 2492416;
static constexpr size_t OFF_SE  = 2493440;           // 2*256
static constexpr size_t OFF_EPART = 2493952;         // 2*256*256
static constexpr size_t OFF_BF16_BYTES = (2493952 + 131072) * 4;  // 32B aligned

// bf16 fragment-buffer offsets (bf16 elements) inside bf16 region
static constexpr size_t FB_EA = 0;       // e_add_w  64x128
static constexpr size_t FB_EM = 8192;    // e_mul_w  64x128
static constexpr size_t FB_EO = 16384;   // e_out_w  128x64
static constexpr size_t FB_F1 = 24576;   // ffE1_w   64x128
static constexpr size_t FB_F2 = 32768;   // ffE2_w   128x64

// ---------------- WMMA helpers ----------------
union FragU { v16bf v; uint4 q[2]; };

__device__ __forceinline__ v8f wmma_bf16(v16bf a, v16bf b, v8f c) {
  return __builtin_amdgcn_wmma_f32_16x16x32_bf16(false, a, false, b, (short)0, c, false, false);
}

// A operand (16xK tile, 16-bit) from LDS, per ISA 7.12.2: lane L holds row M=L%16;
// element e -> K = kk + (L>=16 ? 8:0) + e + (e>=8 ? 8:0)  => two contiguous 16B chunks.
__device__ __forceinline__ v16bf load_a_frag(const __bf16* base, int rowstride, int kk) {
  int lane = threadIdx.x & 31;
  const __bf16* p = base + (lane & 15) * rowstride + kk + ((lane >> 4) << 3);
  FragU u;
  u.q[0] = *(const uint4*)(p);
  u.q[1] = *(const uint4*)(p + 16);
  return u.v;
}

// B operand pre-packed in global memory: frag = [32 lanes][16 elems] contiguous.
__device__ __forceinline__ v16bf load_b_frag(const __bf16* fb, int frag_idx) {
  int lane = threadIdx.x & 31;
  const __bf16* p = fb + (size_t)frag_idx * 512 + lane * 16;
  FragU u;
  u.q[0] = *(const uint4*)(p);
  u.q[1] = *(const uint4*)(p + 8);
  return u.v;
}

// ---------------- cross-lane reductions via ds_swizzle (XOR butterfly) ----------------
// group-of-32 encoding: offset = xor_mask<<10 | or_mask<<5 | and_mask (and=0x1f)
#define SWZF(x, imm) __int_as_float(__builtin_amdgcn_ds_swizzle(__float_as_int(x), (imm)))

__device__ __forceinline__ float redsum16(float x) {
  x += SWZF(x, 0x041F);  // xor 1
  x += SWZF(x, 0x081F);  // xor 2
  x += SWZF(x, 0x101F);  // xor 4
  x += SWZF(x, 0x201F);  // xor 8
  return x;
}
__device__ __forceinline__ float redsum32(float x) {
  x = redsum16(x);
  x += SWZF(x, 0x401F);  // xor 16
  return x;
}
__device__ __forceinline__ float redmax32(float x) {
  x = fmaxf(x, SWZF(x, 0x041F));
  x = fmaxf(x, SWZF(x, 0x081F));
  x = fmaxf(x, SWZF(x, 0x101F));
  x = fmaxf(x, SWZF(x, 0x201F));
  x = fmaxf(x, SWZF(x, 0x401F));
  return x;
}

// ---------------- Tensor Data Mover: 1-D contiguous f32 tile -> LDS ----------------
// D# group0: count=1 | lds_addr | global_addr[56:0] | type=2
// D# group1: data_size=4B; tensor_dim0=tile_dim0=nfloats; tensor_dim1=tile_dim1=1
__device__ __forceinline__ void tdm_load_1d(const float* gsrc, unsigned int lds_off,
                                            unsigned int nfloats) {
  unsigned long long ga = (unsigned long long)(size_t)gsrc;
  u32x4 g0;
  g0[0] = 1u;                                  // count=1
  g0[1] = lds_off;                             // lds_addr (bytes)
  g0[2] = (unsigned int)ga;                    // global_addr[31:0]
  g0[3] = ((unsigned int)(ga >> 32) & 0x01FFFFFFu) | (2u << 30);  // addr[56:32], type=2
  u32x8 g1;
  g1[0] = 2u << 16;                            // data_size=2 (4 bytes); mask=0
  g1[1] = (nfloats & 0xFFFFu) << 16;           // tensor_dim0[15:0]
  g1[2] = (nfloats >> 16) | (1u << 16);        // tensor_dim0[31:16] | tensor_dim1=1
  g1[3] = (nfloats & 0xFFFFu) << 16;           // tile_dim0
  g1[4] = 1u;                                  // tile_dim1=1, tile_dim2=0
  g1[5] = nfloats;                             // tensor_dim0_stride[31:0]
  g1[6] = 0u;
  g1[7] = 0u;
  asm volatile("tensor_load_to_lds %0, %1" :: "s"(g0), "s"(g1) : "memory");
}

// ---------------- kernel 1: pack bf16 weight fragments ----------------
__global__ __launch_bounds__(256) void k_prep_frags(
    const float* ea, const float* em, const float* eo,
    const float* f1, const float* f2, __bf16* dst) {
  int gid = blockIdx.x * 256 + threadIdx.x;        // 0..40959
  if (gid >= 5 * 8192) return;
  int w = gid / 8192, r = gid % 8192;
  const float* src; int Nn;
  if      (w == 0) { src = ea; Nn = 128; }
  else if (w == 1) { src = em; Nn = 128; }
  else if (w == 2) { src = eo; Nn = 64;  }
  else if (w == 3) { src = f1; Nn = 128; }
  else             { src = f2; Nn = 64;  }
  int e = r & 15, L = (r >> 4) & 31, fi = r >> 9;
  int nfN = Nn >> 4;
  int nt = fi % nfN, kki = fi / nfN;
  int K = kki * 32 + ((L >= 16) ? 8 : 0) + e + ((e >= 8) ? 8 : 0);
  int n = nt * 16 + (L & 15);
  dst[(size_t)w * 8192 + r] = (__bf16)src[(size_t)K * Nn + n];
}

// ---------------- kernel 2: y modulation vectors ----------------
__global__ __launch_bounds__(128) void k_ye(
    const float* y, const float* w1, const float* b1, const float* w2, const float* b2,
    const float* w3, const float* b3, const float* w4, const float* b4,
    float* ye1, float* ye2, float* yx1, float* yx2) {
  int b = blockIdx.x, c = threadIdx.x;
  float a1 = b1[c], a2 = b2[c], a3 = b3[c], a4 = b4[c];
  for (int j = 0; j < 64; j++) {
    float v = y[b * 64 + j];
    a1 += v * w1[j * 128 + c]; a2 += v * w2[j * 128 + c];
    a3 += v * w3[j * 128 + c]; a4 += v * w4[j * 128 + c];
  }
  ye1[b * 128 + c] = a1; ye2[b * 128 + c] = a2;
  yx1[b * 128 + c] = a3; yx2[b * 128 + c] = a4;
}

// ---------------- kernel 3: Q/K/V projections ----------------
__global__ __launch_bounds__(128) void k_qkv(
    const float* Xq, const float* Xk,
    const float* qw, const float* qb, const float* kw, const float* kb,
    const float* vw, const float* vb, float* Q, float* K, float* V) {
  int row = blockIdx.x, c = threadIdx.x;
  __shared__ float xq[128], xk[128];
  xq[c] = Xq[(size_t)row * 128 + c];
  xk[c] = Xk[(size_t)row * 128 + c];
  __syncthreads();
  float aq = qb[c], ak = kb[c], av = vb[c];
  for (int j = 0; j < 128; j++) {
    aq += xq[j] * qw[j * 128 + c];
    ak += xk[j] * kw[j * 128 + c];
    av += xk[j] * vw[j * 128 + c];
  }
  Q[(size_t)row * 128 + c] = aq;
  K[(size_t)row * 128 + c] = ak;
  V[(size_t)row * 128 + c] = av;
}

// ---------------- kernel 4: masked stats over X ----------------
__global__ __launch_bounds__(128) void k_statsX(const float* X, const unsigned char* msk, float* out) {
  int b = blockIdx.x, c = threadIdx.x;
  float s = 0.f, ss = 0.f, mn = 3.0e38f, mx = -3.0e38f; int cnt = 0;
  for (int r = 0; r < NB_; r++) {
    if (msk[b * NB_ + r]) {
      float x = X[((size_t)(b * NB_ + r)) * 128 + c];
      s += x; ss += x * x; mn = fminf(mn, x); mx = fmaxf(mx, x); cnt++;
    }
  }
  float denom = (cnt == 0) ? 1.0f : (float)cnt;
  float mean = s / denom;
  float var = (ss + ((float)NB_ - 2.0f * (float)cnt) * mean * mean) / denom;
  float sd = (cnt < 2) ? 0.0f : sqrtf(var + 1e-10f);
  out[b * 512 + c]       = mean;
  out[b * 512 + 128 + c] = sd;
  out[b * 512 + 256 + c] = (cnt == 0) ? 0.0f : mn;
  out[b * 512 + 384 + c] = (cnt == 0) ? 0.0f : mx;
}

// ---------------- kernel 5a/5b: masked stats over E (two-stage) ----------------
__global__ __launch_bounds__(64) void k_statsE_part(
    const float* E, const unsigned char* mq, const unsigned char* mk, float* part) {
  int b = blockIdx.y, blk = blockIdx.x, c = threadIdx.x;
  float s = 0.f, ss = 0.f, mn = 3.0e38f, mx = -3.0e38f;
  for (int i = 0; i < 1024; i++) {
    int row = blk * 1024 + i, q = row >> 9, k = row & 511;
    if (mq[b * NB_ + q] && mk[b * NB_ + k]) {
      float x = E[(((size_t)(b * NB_ + q)) * NB_ + k) * 64 + c];
      s += x; ss += x * x; mn = fminf(mn, x); mx = fmaxf(mx, x);
    }
  }
  size_t o = ((size_t)(b * 256 + blk)) * 256;
  part[o + c] = s; part[o + 64 + c] = ss; part[o + 128 + c] = mn; part[o + 192 + c] = mx;
}

__global__ __launch_bounds__(64) void k_statsE_fin(
    const float* part, const unsigned char* mq, const unsigned char* mk, float* out) {
  int b = blockIdx.x, c = threadIdx.x;
  float s = 0.f, ss = 0.f, mn = 3.0e38f, mx = -3.0e38f;
  for (int p = 0; p < 256; p++) {
    size_t o = ((size_t)(b * 256 + p)) * 256;
    s += part[o + c]; ss += part[o + 64 + c];
    mn = fminf(mn, part[o + 128 + c]); mx = fmaxf(mx, part[o + 192 + c]);
  }
  int cq = 0, ck = 0;
  for (int r = 0; r < NB_; r++) { cq += mq[b * NB_ + r] ? 1 : 0; ck += mk[b * NB_ + r] ? 1 : 0; }
  float cnt = (float)cq * (float)ck;
  float denom = (cnt == 0.f) ? 1.0f : cnt;
  float mean = s / denom;
  float var = (ss + (262144.0f - 2.0f * cnt) * mean * mean) / denom;
  float sd = (cnt < 2.f) ? 0.f : sqrtf(var + 1e-10f);
  out[b * 256 + c]        = mean;
  out[b * 256 + 64 + c]   = sd;
  out[b * 256 + 128 + c]  = (cnt == 0.f) ? 0.f : mn;
  out[b * 256 + 192 + c]  = (cnt == 0.f) ? 0.f : mx;
}

// ---------------- kernel 6: full y pathway -> yo output ----------------
__global__ __launch_bounds__(256) void k_ypath(
    const float* y, const float* sXq, const float* sXk, const float* sE,
    const float* ypw, const float* ypb,
    const float* rqw, const float* rqb, const float* rkw, const float* rkb,
    const float* rew, const float* reb,
    const float* o1w, const float* o1b, const float* o2w, const float* o2b,
    const float* gy1, const float* by1,
    const float* fy1w, const float* fy1b, const float* fy2w, const float* fy2b,
    const float* gy3, const float* by3, float* yout) {
  int b = blockIdx.x, t = threadIdx.x;
  __shared__ float cat[256], h1[64], tv[64], h2[256], tmp[64];
  int seg = t >> 6, i = t & 63;
  float a;
  if      (seg == 0) { a = ypb[i]; for (int j = 0; j < 64;  j++) a += y[b * 64 + j]   * ypw[j * 64 + i]; }
  else if (seg == 1) { a = rqb[i]; for (int j = 0; j < 512; j++) a += sXq[b * 512 + j] * rqw[j * 64 + i]; }
  else if (seg == 2) { a = rkb[i]; for (int j = 0; j < 512; j++) a += sXk[b * 512 + j] * rkw[j * 64 + i]; }
  else               { a = reb[i]; for (int j = 0; j < 256; j++) a += sE[b * 256 + j]  * rew[j * 64 + i]; }
  cat[t] = a;
  __syncthreads();
  if (t < 64) { float v = o1b[t]; for (int j = 0; j < 256; j++) v += cat[j] * o1w[j * 64 + t]; h1[t] = fmaxf(v, 0.f); }
  __syncthreads();
  if (t < 64) { float v = o2b[t]; for (int j = 0; j < 64; j++) v += h1[j] * o2w[j * 64 + t]; tv[t] = y[b * 64 + t] + v; }
  __syncthreads();
  if (t < 64) {
    float s = 0.f, ss = 0.f;
    for (int j = 0; j < 64; j++) { float x = tv[j]; s += x; ss += x * x; }
    float m = s / 64.f, var = ss / 64.f - m * m;
    tmp[t] = (tv[t] - m) * rsqrtf(var + 1e-5f) * gy1[t] + by1[t];
  }
  __syncthreads();
  if (t < 64) tv[t] = tmp[t];
  __syncthreads();
  { float v = fy1b[t]; for (int j = 0; j < 64; j++) v += tv[j] * fy1w[j * 256 + t]; h2[t] = fmaxf(v, 0.f); }
  __syncthreads();
  if (t < 64) { float v = fy2b[t] + tv[t]; for (int j = 0; j < 256; j++) v += h2[j] * fy2w[j * 64 + t]; tmp[t] = v; }
  __syncthreads();
  if (t < 64) {
    float s = 0.f, ss = 0.f;
    for (int j = 0; j < 64; j++) { float x = tmp[j]; s += x; ss += x * x; }
    float m = s / 64.f, var = ss / 64.f - m * m;
    yout[b * 64 + t] = (tmp[t] - m) * rsqrtf(var + 1e-5f) * gy3[t] + by3[t];
  }
}

// ---------------- kernel 7: E-path pass A (WMMA + TDM) ----------------
__global__ __launch_bounds__(32) void k_passA(
    const float* E, const float* Qw, const float* Kw,
    const float* ye1, const float* ye2,
    const __bf16* fa, const __bf16* fm, const __bf16* fo,
    const float* eab, const float* emb, const float* eob,
    const float* g1, const float* bn1,
    const unsigned char* mq, const unsigned char* mk,
    float* Wout, float* EoOut) {
  int lane = threadIdx.x;
  int k0 = blockIdx.x * 16, q = blockIdx.y, b = blockIdx.z;
  __shared__ float  ELDS[16 * 64];
  __shared__ __bf16 Ebf[16 * 64];
  __shared__ float  Ktile[16 * 128];
  __shared__ __bf16 Apost[16 * 128];

  const size_t ebase = (((size_t)(b * NB_ + q)) * NB_ + k0) * 64;

  // Async DMA: 16x64 f32 E tile (contiguous 4KB) -> LDS via Tensor Data Mover.
  tdm_load_1d(E + ebase, (unsigned int)(size_t)&ELDS[0], 16 * 64);
  __builtin_prefetch(E + ebase + 1024, 0, 1);   // next k-tile

  // Meanwhile vector-load the K tile.
  const float4* ksrc = (const float4*)(Kw + ((size_t)(b * NB_ + k0)) * 128);
#pragma unroll
  for (int i = 0; i < 16; i++) {
    float4 v = ksrc[lane + i * 32];
    int fi = (lane + i * 32) * 4;
    Ktile[fi] = v.x; Ktile[fi + 1] = v.y; Ktile[fi + 2] = v.z; Ktile[fi + 3] = v.w;
  }
  __builtin_amdgcn_s_wait_tensorcnt(0);
  // Convert E tile to bf16 alongside the f32 copy.
#pragma unroll
  for (int i = 0; i < 8; i++) {
    int fi = (lane + i * 32) * 4;
    float4 v = *(const float4*)&ELDS[fi];
    Ebf[fi] = (__bf16)v.x; Ebf[fi + 1] = (__bf16)v.y;
    Ebf[fi + 2] = (__bf16)v.z; Ebf[fi + 3] = (__bf16)v.w;
  }
  __syncthreads();

  const int cl = lane & 15;
  const int mo = (lane >> 4) << 3;   // C-layout row offset

  // ---- GEMM1 (E1,E2), attention bias A, head logits, ye-modulated Apost ----
  for (int nt = 0; nt < 8; nt++) {
    v8f a1 = {}; v8f a2 = {};
#pragma unroll
    for (int kki = 0; kki < 2; kki++) {
      v16bf af = load_a_frag(Ebf, 64, kki * 32);
      a1 = wmma_bf16(af, load_b_frag(fa, kki * 8 + nt), a1);
      a2 = wmma_bf16(af, load_b_frag(fm, kki * 8 + nt), a2);
    }
    int c = nt * 16 + cl;
    float qv = Qw[((size_t)(b * NB_ + q)) * 128 + c];
    float b1v = eab[c], b2v = emb[c];
    float yav = ye1[b * 128 + c], ymv = ye2[b * 128 + c];
    float lg[8];
#pragma unroll
    for (int v = 0; v < 8; v++) {
      int m = v + mo;
      float Av = (a1[v] + b1v) + (a2[v] + b2v + 1.0f) * qv * Ktile[m * 128 + c] * 0.25f;
      lg[v] = redsum16(Av);
      Apost[m * 128 + c] = (__bf16)(yav + (ymv + 1.0f) * Av);
    }
    if (cl == 0) {
      size_t wb = (((size_t)(b * NB_ + q)) * NB_ + k0 + mo) * 8 + nt;
#pragma unroll
      for (int v = 0; v < 8; v++) Wout[wb + (size_t)v * 8] = lg[v];
    }
  }
  __syncthreads();

  // ---- GEMM2 (e_out), mask, residual ----
  float tpre[4][8];
  for (int nt = 0; nt < 4; nt++) {
    v8f acc = {};
#pragma unroll
    for (int kki = 0; kki < 4; kki++)
      acc = wmma_bf16(load_a_frag(Apost, 128, kki * 32), load_b_frag(fo, kki * 4 + nt), acc);
    int c = nt * 16 + cl;
    float ob = eob[c];
    unsigned char mqv = mq[b * NB_ + q];
#pragma unroll
    for (int v = 0; v < 8; v++) {
      int m = v + mo;
      float msk = (mqv && mk[b * NB_ + k0 + m]) ? 1.0f : 0.0f;
      tpre[nt][v] = ELDS[m * 64 + c] + msk * (acc[v] + ob);
    }
  }
  // ---- layernorm (normE1) per row, write t1 to d_out E region ----
  float mu[8], ri[8];
#pragma unroll
  for (int v = 0; v < 8; v++) {
    float s = tpre[0][v] + tpre[1][v] + tpre[2][v] + tpre[3][v];
    float ss = tpre[0][v] * tpre[0][v] + tpre[1][v] * tpre[1][v] +
               tpre[2][v] * tpre[2][v] + tpre[3][v] * tpre[3][v];
    s = redsum16(s); ss = redsum16(ss);
    float m_ = s * (1.0f / 64.0f);
    float var = ss * (1.0f / 64.0f) - m_ * m_;
    mu[v] = m_; ri[v] = rsqrtf(var + 1e-5f);
  }
  for (int nt = 0; nt < 4; nt++) {
    int c = nt * 16 + cl;
    float gg = g1[c], bb = bn1[c];
#pragma unroll
    for (int v = 0; v < 8; v++) {
      int m = v + mo;
      EoOut[ebase + (size_t)m * 64 + c] = (tpre[nt][v] - mu[v]) * ri[v] * gg + bb;
    }
  }
}

// ---------------- kernel 8: E feed-forward epilogue (WMMA + TDM, in-place) ----------------
__global__ __launch_bounds__(32) void k_effn(
    float* Eo, const __bf16* f1, const __bf16* f2,
    const float* b1, const float* b2, const float* g3, const float* bn3) {
  int lane = threadIdx.x;
  int k0 = blockIdx.x * 16, q = blockIdx.y, b = blockIdx.z;
  __shared__ float  T[16 * 64];
  __shared__ __bf16 Tbf[16 * 64];
  __shared__ __bf16 Hb[16 * 128];
  const size_t ebase = (((size_t)(b * NB_ + q)) * NB_ + k0) * 64;

  tdm_load_1d(Eo + ebase, (unsigned int)(size_t)&T[0], 16 * 64);
  __builtin_amdgcn_s_wait_tensorcnt(0);
#pragma unroll
  for (int i = 0; i < 8; i++) {
    int fi = (lane + i * 32) * 4;
    float4 v = *(const float4*)&T[fi];
    Tbf[fi] = (__bf16)v.x; Tbf[fi + 1] = (__bf16)v.y;
    Tbf[fi + 2] = (__bf16)v.z; Tbf[fi + 3] = (__bf16)v.w;
  }
  __syncthreads();
  const int cl = lane & 15;
  const int mo = (lane >> 4) << 3;

  for (int nt = 0; nt < 8; nt++) {
    v8f acc = {};
#pragma unroll
    for (int kki = 0; kki < 2; kki++)
      acc = wmma_bf16(load_a_frag(Tbf, 64, kki * 32), load_b_frag(f1, kki * 8 + nt), acc);
    int c = nt * 16 + cl;
    float bb = b1[c];
#pragma unroll
    for (int v = 0; v < 8; v++) {
      int m = v + mo;
      Hb[m * 128 + c] = (__bf16)fmaxf(acc[v] + bb, 0.0f);
    }
  }
  __syncthreads();

  float tpre[4][8];
  for (int nt = 0; nt < 4; nt++) {
    v8f acc = {};
#pragma unroll
    for (int kki = 0; kki < 4; kki++)
      acc = wmma_bf16(load_a_frag(Hb, 128, kki * 32), load_b_frag(f2, kki * 4 + nt), acc);
    int c = nt * 16 + cl;
    float bb = b2[c];
#pragma unroll
    for (int v = 0; v < 8; v++) {
      int m = v + mo;
      tpre[nt][v] = T[m * 64 + c] + acc[v] + bb;
    }
  }
  float mu[8], ri[8];
#pragma unroll
  for (int v = 0; v < 8; v++) {
    float s = tpre[0][v] + tpre[1][v] + tpre[2][v] + tpre[3][v];
    float ss = tpre[0][v] * tpre[0][v] + tpre[1][v] * tpre[1][v] +
               tpre[2][v] * tpre[2][v] + tpre[3][v] * tpre[3][v];
    s = redsum16(s); ss = redsum16(ss);
    float m_ = s * (1.0f / 64.0f);
    float var = ss * (1.0f / 64.0f) - m_ * m_;
    mu[v] = m_; ri[v] = rsqrtf(var + 1e-5f);
  }
  for (int nt = 0; nt < 4; nt++) {
    int c = nt * 16 + cl;
    float gg = g3[c], bb = bn3[c];
#pragma unroll
    for (int v = 0; v < 8; v++) {
      int m = v + mo;
      Eo[ebase + (size_t)m * 64 + c] = (tpre[nt][v] - mu[v]) * ri[v] * gg + bb;
    }
  }
}

// ---------------- kernel 9: softmax, attn*V, full X epilogue ----------------
__global__ __launch_bounds__(256) void k_attnX(
    const float* Wlog, const float* V, const float* Xq,
    const float* yx1, const float* yx2,
    const float* xow, const float* xob,
    const float* fx1w, const float* fx1b, const float* fx2w, const float* fx2b,
    const float* gX1, const float* bX1, const float* gX3, const float* bX3,
    const unsigned char* mq, const unsigned char* mk, float* Xout) {
  int q = blockIdx.x & 511, b = blockIdx.x >> 9;
  int t = threadIdx.x, lane = t & 31, hd = t >> 5;
  __shared__ float attn[8 * 512];
  __shared__ float xrow[128], X1[128], tX[128], hbuf[512];
  const size_t wbase = (((size_t)(b * NB_ + q)) * NB_) * 8;
  float mx = -3.0e38f;
  for (int it = 0; it < 16; it++) {
    int k = lane + it * 32;
    float w = Wlog[wbase + (size_t)k * 8 + hd];
    float s = (mk[b * NB_ + k] != 0) ? w : -3.0e38f;
    attn[hd * 512 + k] = s;
    mx = fmaxf(mx, s);
  }
  mx = redmax32(mx);
  float sm = 0.f;
  for (int it = 0; it < 16; it++) {
    int k = lane + it * 32;
    float e = (mk[b * NB_ + k] != 0) ? __expf(attn[hd * 512 + k] - mx) : 0.0f;
    attn[hd * 512 + k] = e;
    sm += e;
  }
  sm = redsum32(sm);
  float inv = 1.0f / fmaxf(sm, 1e-30f);
  for (int it = 0; it < 16; it++) {
    int k = lane + it * 32;
    attn[hd * 512 + k] *= inv;
  }
  __syncthreads();
  if (t < 128) {
    int h2 = t >> 4;
    float acc = 0.f;
    for (int k = 0; k < 512; k++)
      acc += attn[h2 * 512 + k] * V[((size_t)(b * NB_ + k)) * 128 + t];
    xrow[t] = yx1[b * 128 + t] + (yx2[b * 128 + t] + 1.0f) * acc;
  }
  __syncthreads();
  if (t < 128) {
    float nx = xob[t];
    for (int j = 0; j < 128; j++) nx += xrow[j] * xow[j * 128 + t];
    if (!mq[b * NB_ + q]) nx = 0.0f;
    tX[t] = Xq[((size_t)(b * NB_ + q)) * 128 + t] + nx;
  }
  __syncthreads();
  if (t < 128) {
    float s = 0.f, ss = 0.f;
    for (int j = 0; j < 128; j++) { float x = tX[j]; s += x; ss += x * x; }
    float m = s / 128.f, var = ss / 128.f - m * m;
    X1[t] = (tX[t] - m) * rsqrtf(var + 1e-5f) * gX1[t] + bX1[t];
  }
  __syncthreads();
  for (int jj = t; jj < 512; jj += 256) {
    float hv = fx1b[jj];
    for (int c2 = 0; c2 < 128; c2++) hv += X1[c2] * fx1w[c2 * 512 + jj];
    hbuf[jj] = fmaxf(hv, 0.0f);
  }
  __syncthreads();
  if (t < 128) {
    float o = fx2b[t] + X1[t];
    for (int j = 0; j < 512; j++) o += hbuf[j] * fx2w[j * 128 + t];
    tX[t] = o;
  }
  __syncthreads();
  if (t < 128) {
    float s = 0.f, ss = 0.f;
    for (int j = 0; j < 128; j++) { float x = tX[j]; s += x; ss += x * x; }
    float m = s / 128.f, var = ss / 128.f - m * m;
    Xout[((size_t)(b * NB_ + q)) * 128 + t] = (tX[t] - m) * rsqrtf(var + 1e-5f) * gX3[t] + bX3[t];
  }
}

// ---------------- launcher ----------------
extern "C" void kernel_launch(void* const* d_in, const int* in_sizes, int n_in,
                              void* d_out, int out_size, void* d_ws, size_t ws_size,
                              hipStream_t stream) {
  const float* Xq = (const float*)d_in[0];
  const float* Xk = (const float*)d_in[1];
  const float* E  = (const float*)d_in[2];
  const float* y  = (const float*)d_in[3];
  const unsigned char* mq = (const unsigned char*)d_in[4];
  const unsigned char* mk = (const unsigned char*)d_in[5];
  const float* q_w = (const float*)d_in[6];   const float* q_b = (const float*)d_in[7];
  const float* k_w = (const float*)d_in[8];   const float* k_b = (const float*)d_in[9];
  const float* v_w = (const float*)d_in[10];  const float* v_b = (const float*)d_in[11];
  const float* xo_w = (const float*)d_in[12]; const float* xo_b = (const float*)d_in[13];
  const float* ea_w = (const float*)d_in[14]; const float* ea_b = (const float*)d_in[15];
  const float* em_w = (const float*)d_in[16]; const float* em_b = (const float*)d_in[17];
  const float* yea_w = (const float*)d_in[18]; const float* yea_b = (const float*)d_in[19];
  const float* yem_w = (const float*)d_in[20]; const float* yem_b = (const float*)d_in[21];
  const float* yxa_w = (const float*)d_in[22]; const float* yxa_b = (const float*)d_in[23];
  const float* yxm_w = (const float*)d_in[24]; const float* yxm_b = (const float*)d_in[25];
  const float* yp_w = (const float*)d_in[26]; const float* yp_b = (const float*)d_in[27];
  const float* eo_w = (const float*)d_in[28]; const float* eo_b = (const float*)d_in[29];
  const float* re_w = (const float*)d_in[30]; const float* re_b = (const float*)d_in[31];
  const float* rq_w = (const float*)d_in[32]; const float* rq_b = (const float*)d_in[33];
  const float* rk_w = (const float*)d_in[34]; const float* rk_b = (const float*)d_in[35];
  const float* o1_w = (const float*)d_in[36]; const float* o1_b = (const float*)d_in[37];
  const float* o2_w = (const float*)d_in[38]; const float* o2_b = (const float*)d_in[39];
  const float* fx1_w = (const float*)d_in[40]; const float* fx1_b = (const float*)d_in[41];
  const float* fx2_w = (const float*)d_in[42]; const float* fx2_b = (const float*)d_in[43];
  const float* fe1_w = (const float*)d_in[44]; const float* fe1_b = (const float*)d_in[45];
  const float* fe2_w = (const float*)d_in[46]; const float* fe2_b = (const float*)d_in[47];
  const float* fy1_w = (const float*)d_in[48]; const float* fy1_b = (const float*)d_in[49];
  const float* fy2_w = (const float*)d_in[50]; const float* fy2_b = (const float*)d_in[51];
  const float* gX1 = (const float*)d_in[52]; const float* bX1 = (const float*)d_in[53];
  const float* gE1 = (const float*)d_in[54]; const float* bE1 = (const float*)d_in[55];
  const float* gy1 = (const float*)d_in[56]; const float* by1 = (const float*)d_in[57];
  const float* gX3 = (const float*)d_in[58]; const float* bX3 = (const float*)d_in[59];
  const float* gE3 = (const float*)d_in[60]; const float* bE3 = (const float*)d_in[61];
  const float* gy3 = (const float*)d_in[62]; const float* by3 = (const float*)d_in[63];

  float* ws = (float*)d_ws;
  float* Q   = ws + OFF_Q;
  float* K   = ws + OFF_K;
  float* V   = ws + OFF_V;
  float* Wlg = ws + OFF_W;
  float* ye1 = ws + OFF_YE1; float* ye2 = ws + OFF_YE2;
  float* yx1 = ws + OFF_YX1; float* yx2 = ws + OFF_YX2;
  float* sXq = ws + OFF_SXQ; float* sXk = ws + OFF_SXK;
  float* sE  = ws + OFF_SE;  float* epart = ws + OFF_EPART;
  __bf16* fbase = (__bf16*)((char*)d_ws + OFF_BF16_BYTES);
  __bf16* fEA = fbase + FB_EA; __bf16* fEM = fbase + FB_EM;
  __bf16* fEO = fbase + FB_EO; __bf16* fF1 = fbase + FB_F1; __bf16* fF2 = fbase + FB_F2;

  float* Xout  = (float*)d_out;
  float* EoOut = (float*)d_out + 131072;
  float* yOut  = (float*)d_out + 131072 + 33554432;

  k_prep_frags<<<160, 256, 0, stream>>>(ea_w, em_w, eo_w, fe1_w, fe2_w, fbase);
  k_ye<<<2, 128, 0, stream>>>(y, yea_w, yea_b, yem_w, yem_b, yxa_w, yxa_b, yxm_w, yxm_b,
                              ye1, ye2, yx1, yx2);
  k_qkv<<<1024, 128, 0, stream>>>(Xq, Xk, q_w, q_b, k_w, k_b, v_w, v_b, Q, K, V);
  k_statsX<<<2, 128, 0, stream>>>(Xq, mq, sXq);
  k_statsX<<<2, 128, 0, stream>>>(Xk, mk, sXk);
  k_statsE_part<<<dim3(256, 2), 64, 0, stream>>>(E, mq, mk, epart);
  k_statsE_fin<<<2, 64, 0, stream>>>(epart, mq, mk, sE);
  k_ypath<<<2, 256, 0, stream>>>(y, sXq, sXk, sE, yp_w, yp_b, rq_w, rq_b, rk_w, rk_b,
                                 re_w, re_b, o1_w, o1_b, o2_w, o2_b, gy1, by1,
                                 fy1_w, fy1_b, fy2_w, fy2_b, gy3, by3, yOut);
  k_passA<<<dim3(32, 512, 2), 32, 0, stream>>>(E, Q, K, ye1, ye2, fEA, fEM, fEO,
                                               ea_b, em_b, eo_b, gE1, bE1, mq, mk,
                                               Wlg, EoOut);
  k_effn<<<dim3(32, 512, 2), 32, 0, stream>>>(EoOut, fF1, fF2, fe1_b, fe2_b, gE3, bE3);
  k_attnX<<<1024, 256, 0, stream>>>(Wlg, V, Xq, yx1, yx2, xo_w, xo_b,
                                    fx1_w, fx1_b, fx2_w, fx2_b,
                                    gX1, bX1, gX3, bX3, mq, mk, Xout);
}
